// LatentNODE_86157043958233
// MI455X (gfx1250) — matrix operand order repaired and checked by reading, not compiled, not run on hardware
//
#include <hip/hip_runtime.h>
#include <cstddef>

// ---------------- problem constants ----------------
#define B_TOT    4096
#define T_SEQ    64
#define F_IN     32
#define C_CTY    64
#define L_LAT    256
#define W_HID    512
#define NT_OUT   8
#define H_HOR    12
#define NSUB     10
#define BT       16            // batch rows per block
#define NTHREADS 256           // 8 wave32s
#define NEL      (BT * L_LAT)  // 4096 state elements per block

// ---------------- vector types for WMMA ----------------
typedef __attribute__((ext_vector_type(16))) __bf16    v16bf;
typedef __attribute__((ext_vector_type(8)))  float     v8f;
typedef __attribute__((ext_vector_type(4)))  unsigned  u32x4;

union FragU { u32x4 q[2]; v16bf v; };

__device__ __forceinline__ unsigned short f2bf(float f) {
  union { float f; unsigned u; } v; v.f = f;
  unsigned u = v.u;
  return (unsigned short)((u + 0x7FFFu + ((u >> 16) & 1u)) >> 16); // RNE
}
__device__ __forceinline__ float sigmf(float x) { return 1.f / (1.f + __expf(-x)); }
__device__ __forceinline__ float eluf (float x) { return x > 0.f ? x : (__expf(x) - 1.f); }

// A-fragment (16xK tile, bf16) from row-major LDS buffer with row stride `stride` halfs.
// Contiguous k order (matches B packing): lanes 0-15 -> M=0..15, K in {0..15}; lanes 16-31 -> K in {16..31}.
__device__ __forceinline__ v16bf load_afrag(const unsigned short* lds, int stride,
                                            int kbase, int lane) {
  int m  = lane & 15;
  int ko = (lane >> 4) * 16;
  const unsigned short* p = lds + m * stride + kbase + ko;
  FragU f;
  f.q[0] = *(const u32x4*)(p);
  f.q[1] = *(const u32x4*)(p + 8);
  return f.v;
}

// B-fragment from fragment-linear packed global buffer: frag fi = 32 lanes x 32 bytes.
__device__ __forceinline__ v16bf load_bfrag(const unsigned short* g, int fi, int lane) {
  const u32x4* p = (const u32x4*)(g + (size_t)fi * 512 + lane * 16);
  FragU f;
  f.q[0] = p[0];
  f.q[1] = p[1];
  return f.v;
}

// One 16x16 output tile of A[16,K] x B[K,16*ntl] (column block nt), K = 32*ktiles.
__device__ __forceinline__ v8f gemm_tile(const unsigned short* a_lds, int astride,
                                         const unsigned short* b_pack,
                                         int ktiles, int ntl, int nt, int lane) {
  v8f acc = {0.f, 0.f, 0.f, 0.f, 0.f, 0.f, 0.f, 0.f};
  for (int kt = 0; kt < ktiles; ++kt) {
    v16bf a = load_afrag(a_lds, astride, kt * 32, lane);
    v16bf b = load_bfrag(b_pack, kt * ntl + nt, lane);
    acc = __builtin_amdgcn_wmma_f32_16x16x32_bf16(false, a, false, b,
                                                  (short)0, acc, false, false);
  }
  return acc;
}

// ---------------- weight packing kernel ----------------
// Packs B = W^T (so B[k][n] = W[n*srcStride + k]) in bf16, fragment-linear order:
// frag fi = kt*(Ntot/16)+nt, within frag lane l stores 16 halfs:
// n = nt*16 + (l&15), K = kt*32 + (l>>4)*16 + j, j=0..15  (2 halfs per dword, K even low)
__global__ void pack_weights(const float* __restrict__ W, unsigned short* __restrict__ dst,
                             int Ktot, int Ntot, int srcStride, int total) {
  int gid = blockIdx.x * blockDim.x + threadIdx.x;
  if (gid >= total) return;
  int fi = gid >> 5, l = gid & 31;
  int ntl = Ntot >> 4;
  int kt = fi / ntl, nt = fi - kt * ntl;
  int n  = nt * 16 + (l & 15);
  int k0 = kt * 32 + ((l >> 4) << 4);
  const float* src = W + (size_t)n * srcStride + k0;
  union { unsigned short h[16]; u32x4 q[2]; } u;
#pragma unroll
  for (int j = 0; j < 16; ++j) u.h[j] = f2bf(src[j]);
  u32x4* d = (u32x4*)(dst + (size_t)fi * 512 + l * 16);
  d[0] = u.q[0];
  d[1] = u.q[1];
}

// ---------------- persistent main kernel ----------------
// Each block owns BT=16 batch rows end-to-end: GRU encode -> Tsit5 ODE -> decode.
__global__ __launch_bounds__(NTHREADS, 1)
void lnode_main(const float* __restrict__ x_seq,
                const int*   __restrict__ cidx,
                const float* __restrict__ W_ih,
                const float* __restrict__ b_ih,
                const float* __restrict__ b_n,
                const float* __restrict__ b0v,
                const float* __restrict__ b1v,
                const float* __restrict__ b2v,
                const float* __restrict__ W_dec,
                const float* __restrict__ b_dec,
                const unsigned short* __restrict__ wf0,
                const unsigned short* __restrict__ wf1,
                const unsigned short* __restrict__ wf2,
                const unsigned short* __restrict__ wfih,
                const unsigned short* __restrict__ wfhh,
                float* __restrict__ out) {
  extern __shared__ char smem[];
  float*          y    = (float*)(smem);                      // 16x256 f32   (16KB)
  float*          kb0  = (float*)(smem + 16384);              // 6 x 16x256   (96KB)
  unsigned short* ysbf = (unsigned short*)(smem + 114688);    // 16x256 bf16  (8KB)
  unsigned short* a0bf = (unsigned short*)(smem + 122880);    // 16x512 bf16  (16KB)
  unsigned short* a1bf = (unsigned short*)(smem + 139264);    // 16x512 bf16  (16KB)
  unsigned short* xsbf = (unsigned short*)(smem + 155648);    // 16x32  bf16  (1KB)

  float* kb[6];
#pragma unroll
  for (int i = 0; i < 6; ++i) kb[i] = kb0 + i * NEL;

  // GRU-phase aliases (dead during ODE phase)
  float* cb   = kb[0];           // 16x768 f32 country+b_ih bias   (= kb0..kb2)
  float* sbuf = kb[3];           // 16x512 f32 combined r,z pre-act (= kb3..kb4)
  float* innb = kb[5];           // 16x256 f32 input-path n pre-act
  float* hnb  = (float*)a0bf;    // 16x256 f32 hidden-path n pre-act

  const int tid  = threadIdx.x;
  const int lane = tid & 31;
  const int wid  = tid >> 5;
  const int lrow = (lane >> 4) * 8;   // C/D layout: VGPR i -> row i + 8*(lane/16)
  const int lcol = lane & 15;         // C/D layout: col = lane % 16
  const int rowbase = blockIdx.x * BT;

  // ================= GRU encoder =================
  // one-hot @ W_ih^T is a row gather -> fold into per-row bias cb (+ b_ih)
  for (int e = tid; e < BT * 768; e += NTHREADS) {
    int m = e / 768, j = e - m * 768;
    int c = cidx[rowbase + m];
    cb[e] = W_ih[(size_t)j * (F_IN + C_CTY) + F_IN + c] + b_ih[j];
  }
  for (int e = tid; e < NEL; e += NTHREADS) y[e] = 0.f;
  __syncthreads();

  for (int t = 0; t < T_SEQ; ++t) {
    // stage x_t and h as bf16
    for (int e = tid; e < BT * F_IN; e += NTHREADS) {
      int m = e >> 5, f = e & 31;
      xsbf[e] = f2bf(x_seq[(size_t)(rowbase + m) * (T_SEQ * F_IN) + t * F_IN + f]);
    }
    for (int e = tid; e < NEL; e += NTHREADS) ysbf[e] = f2bf(y[e]);
    __syncthreads();

    // 48 N-tiles of the 768-wide gate pre-activations, 6 per wave
    for (int i = 0; i < 6; ++i) {
      int nt = wid * 6 + i;
      v8f ah = gemm_tile(ysbf, L_LAT, wfhh, 8, 48, nt, lane);  // h  @ W_hh^T
      v8f ax = gemm_tile(xsbf, F_IN, wfih, 1, 48, nt, lane);   // x_t@ W_ih[:,:32]^T
      int n = nt * 16 + lcol;
      if (n < 512) {  // r,z ranges: input+hidden combine additively
#pragma unroll
        for (int r = 0; r < 8; ++r)
          sbuf[(lrow + r) * 512 + n] = ax[r] + ah[r] + cb[(lrow + r) * 768 + n];
      } else {        // n range: keep input and hidden paths separate
        int nl = n - 512;
#pragma unroll
        for (int r = 0; r < 8; ++r) {
          innb[(lrow + r) * 256 + nl] = ax[r] + cb[(lrow + r) * 768 + n];
          hnb [(lrow + r) * 256 + nl] = ah[r];
        }
      }
    }
    __syncthreads();

    // gate elementwise update
    for (int e = tid; e < NEL; e += NTHREADS) {
      int m = e >> 8, c = e & 255;
      float rr = sigmf(sbuf[m * 512 + c]);
      float zz = sigmf(sbuf[m * 512 + 256 + c]);
      float nn = tanhf(innb[e] + rr * (hnb[e] + b_n[c]));
      float ho = y[e];
      y[e] = nn + zz * (ho - nn);
    }
    __syncthreads();
  }
  // y now holds h_final == ODE initial state.

  // ================= Tsit5 neural ODE =================
  const float dt = 1.0f / (float)NSUB;
  const float C2[1] = {0.161f};
  const float C3[2] = {-0.008480655492356989f, 0.335480655492357f};
  const float C4[3] = {2.8971530571054935f, -6.359448489975075f, 4.362295432869581f};
  const float C5[4] = {5.325864828439257f, -11.748883564062828f, 7.4955393428898365f,
                       -0.09249506636175525f};
  const float C6[5] = {5.86145544294642f, -12.92096931784711f, 8.159367898576159f,
                       -0.071584973281401f, -0.028269050394068383f};
  const float BC[6] = {0.09646076681806523f, 0.01f, 0.4798896504144996f,
                       1.379008574103742f, -3.290069515436081f, 2.324710524099774f};

  // ELU MLP f(y): 256 -> 512 -> 512 -> 256, all WMMA bf16 with f32 accumulate
  auto mlp = [&](const unsigned short* abf, float* kout) {
    // layer 0: [16,256] x [256,512]
    for (int i = 0; i < 4; ++i) {
      int nt = wid * 4 + i;
      v8f acc = gemm_tile(abf, L_LAT, wf0, 8, 32, nt, lane);
      int n = nt * 16 + lcol;
      float bias = b0v[n];
#pragma unroll
      for (int r = 0; r < 8; ++r)
        a0bf[(lrow + r) * W_HID + n] = f2bf(eluf(acc[r] + bias));
    }
    __syncthreads();
    // layer 1: [16,512] x [512,512]
    for (int i = 0; i < 4; ++i) {
      int nt = wid * 4 + i;
      v8f acc = gemm_tile(a0bf, W_HID, wf1, 16, 32, nt, lane);
      int n = nt * 16 + lcol;
      float bias = b1v[n];
#pragma unroll
      for (int r = 0; r < 8; ++r)
        a1bf[(lrow + r) * W_HID + n] = f2bf(eluf(acc[r] + bias));
    }
    __syncthreads();
    // layer 2: [16,512] x [512,256]
    for (int i = 0; i < 2; ++i) {
      int nt = wid * 2 + i;
      v8f acc = gemm_tile(a1bf, W_HID, wf2, 16, 16, nt, lane);
      int n = nt * 16 + lcol;
      float bias = b2v[n];
#pragma unroll
      for (int r = 0; r < 8; ++r)
        kout[(lrow + r) * L_LAT + n] = acc[r] + bias;
    }
    __syncthreads();
  };

  // stage input: ysbf = bf16( y + dt * sum_j cf[j] * k_j )
  auto combine = [&](const float* cf, int nk) {
    for (int e = tid; e < NEL; e += NTHREADS) {
      float s = y[e];
      for (int j = 0; j < nk; ++j) s += dt * cf[j] * kb[j][e];
      ysbf[e] = f2bf(s);
    }
    __syncthreads();
  };

  for (int u = 0; u < H_HOR; ++u) {
    for (int s = 0; s < NSUB; ++s) {
      combine(nullptr, 0); mlp(ysbf, kb[0]);   // k1 = f(y)
      combine(C2, 1);      mlp(ysbf, kb[1]);
      combine(C3, 2);      mlp(ysbf, kb[2]);
      combine(C4, 3);      mlp(ysbf, kb[3]);
      combine(C5, 4);      mlp(ysbf, kb[4]);
      combine(C6, 5);      mlp(ysbf, kb[5]);
      for (int e = tid; e < NEL; e += NTHREADS) {
        float acc = 0.f;
#pragma unroll
        for (int j = 0; j < 6; ++j) acc += BC[j] * kb[j][e];
        y[e] += dt * acc;
      }
      __syncthreads();
    }
    // ---- decode this horizon: out[b, u, t] = y[b,:] . W_dec[t,:] + b_dec[t] ----
    if (tid < BT * NT_OUT) {
      int m = tid >> 3, t8 = tid & 7;
      float s = b_dec[t8];
      const float* wd = W_dec + t8 * L_LAT;
      const float* yr = y + m * L_LAT;
      for (int c = 0; c < L_LAT; ++c) s += yr[c] * wd[c];
      out[(size_t)(rowbase + m) * (H_HOR * NT_OUT) + u * NT_OUT + t8] = s;
    }
    __syncthreads();
  }
}

// ---------------- host launch ----------------
extern "C" void kernel_launch(void* const* d_in, const int* in_sizes, int n_in,
                              void* d_out, int out_size, void* d_ws, size_t ws_size,
                              hipStream_t stream) {
  (void)in_sizes; (void)n_in; (void)out_size; (void)ws_size;
  const float* x_seq = (const float*)d_in[0];
  const int*   cidx  = (const int*)d_in[1];
  // d_in[2] = horizon (compile-time H_HOR=12)
  const float* W_ih  = (const float*)d_in[3];
  const float* W_hh  = (const float*)d_in[4];
  const float* b_ih  = (const float*)d_in[5];
  const float* b_n   = (const float*)d_in[6];
  const float* W0    = (const float*)d_in[7];
  const float* b0    = (const float*)d_in[8];
  const float* W1    = (const float*)d_in[9];
  const float* b1    = (const float*)d_in[10];
  const float* W2    = (const float*)d_in[11];
  const float* b2    = (const float*)d_in[12];
  const float* W_dec = (const float*)d_in[13];
  const float* b_dec = (const float*)d_in[14];
  float* out = (float*)d_out;

  // packed bf16 weight buffers in workspace (fragment-linear), ~1.5 MB total
  unsigned short* wf0  = (unsigned short*)d_ws;            // 256 frags
  unsigned short* wf1  = wf0  + (size_t)256 * 512;         // 512 frags
  unsigned short* wf2  = wf1  + (size_t)512 * 512;         // 256 frags
  unsigned short* wfih = wf2  + (size_t)256 * 512;         //  48 frags
  unsigned short* wfhh = wfih + (size_t)48  * 512;         // 384 frags

  auto packLaunch = [&](const float* W, unsigned short* dst, int Ktot, int Ntot,
                        int srcStride) {
    int frags  = (Ktot / 32) * (Ntot / 16);
    int total  = frags * 32;
    int blocks = (total + NTHREADS - 1) / NTHREADS;
    pack_weights<<<blocks, NTHREADS, 0, stream>>>(W, dst, Ktot, Ntot, srcStride, total);
  };
  packLaunch(W0,   wf0,  256, 512, 256);  // B0 = W0^T  [256 x 512]
  packLaunch(W1,   wf1,  512, 512, 512);  // B1 = W1^T  [512 x 512]
  packLaunch(W2,   wf2,  512, 256, 512);  // B2 = W2^T  [512 x 256]
  packLaunch(W_ih, wfih, 32,  768, 96);   // Bih = W_ih[:, :32]^T [32 x 768]
  packLaunch(W_hh, wfhh, 256, 768, 256);  // Bhh = W_hh^T [256 x 768]

  constexpr int SMEM_BYTES = 16384 + 6 * 16384 + 8192 + 16384 + 16384 + 1024; // 156672
  hipFuncSetAttribute(reinterpret_cast<const void*>(lnode_main),
                      hipFuncAttributeMaxDynamicSharedMemorySize, SMEM_BYTES);
  lnode_main<<<B_TOT / BT, NTHREADS, SMEM_BYTES, stream>>>(
      x_seq, cidx, W_ih, b_ih, b_n, b0, b1, b2, W_dec, b_dec,
      wf0, wf1, wf2, wfih, wfhh, out);
}
